// hierarchically_modular_shared_modules_mlp_35321811042832
// MI455X (gfx1250) — compile-verified, matrix operand order checked by reading, not run on hardware
//
#include <hip/hip_runtime.h>
#include <cstddef>
#include <cstdint>

typedef __attribute__((ext_vector_type(16))) __bf16 v16bf;
typedef __attribute__((ext_vector_type(8)))  float  v8f;

#define TB   128   // rows per block
#define LDA  34    // Xa K-pitch (even -> dword-pair aligned)
#define LDW  40    // Wt K-pitch (80B rows -> 16B aligned for async b128 stores)
#define LDY  258   // Y1/Y2 N-pitch (even)
#define K1PAD 800  // W1 K padded to 25 chunks of 32

// ---------------------------------------------------------------------------
// CDNA5 async global->LDS copy (ASYNCcnt-tracked) + waits
// ---------------------------------------------------------------------------
__device__ __forceinline__ void async_b128(uint32_t lds_byte_off, const void* gsrc) {
  asm volatile("global_load_async_to_lds_b128 %0, %1, off"
               :: "v"(lds_byte_off), "v"((unsigned long long)(uintptr_t)gsrc)
               : "memory");
}
__device__ __forceinline__ void wait_async4() { asm volatile("s_wait_asynccnt 0x4" ::: "memory"); }
__device__ __forceinline__ void wait_async0() { asm volatile("s_wait_asynccnt 0x0" ::: "memory"); }

// ---------------------------------------------------------------------------
// Kernel 1: routing decisions (argmax / top-2 of tiny embeddings)
// R layout: [0..3]=c_s img, [4..7]=m_s img, [8..10]=L1 top1, [11..13]=L1 top2,
// [14..16]=L1 loc, [17..18]=L2 top1, [19..20]=L2 top2, [21..22]=L2 loc,
// [23]=L3 top1, [24]=L3 top2
// ---------------------------------------------------------------------------
__global__ void route_kernel(const float* __restrict__ e0, const float* __restrict__ l0,
                             const float* __restrict__ e1, const float* __restrict__ l1,
                             const float* __restrict__ e2, const float* __restrict__ l2,
                             const float* __restrict__ e3, int* __restrict__ R) {
  if (threadIdx.x != 0) return;
  for (int s = 0; s < 4; ++s) {
    int bc = 0; float bv = e0[s];
    for (int c = 1; c < 8; ++c) { float v = e0[c * 4 + s]; if (v > bv) { bv = v; bc = c; } }
    R[s] = bc;
    int bm = 0; float bmv = l0[s];
    for (int m = 1; m < 8; ++m) { float v = l0[m * 4 + s]; if (v > bmv) { bmv = v; bm = m; } }
    R[4 + s] = bm;
  }
  for (int s = 0; s < 3; ++s) {
    int i1 = 0; float v1 = e1[s];
    for (int d = 1; d < 12; ++d) { float v = e1[d * 3 + s]; if (v > v1) { v1 = v; i1 = d; } }
    int i2 = -1; float v2 = -1e30f;
    for (int d = 0; d < 12; ++d) { if (d == i1) continue; float v = e1[d * 3 + s]; if (v > v2) { v2 = v; i2 = d; } }
    R[8 + s] = i1; R[11 + s] = i2;
    R[14 + s] = (l1[s] >= l1[3 + s]) ? 0 : 1;
  }
  for (int s = 0; s < 2; ++s) {
    int i1 = 0; float v1 = e2[s];
    for (int d = 1; d < 3; ++d) { float v = e2[d * 2 + s]; if (v > v1) { v1 = v; i1 = d; } }
    int i2 = -1; float v2 = -1e30f;
    for (int d = 0; d < 3; ++d) { if (d == i1) continue; float v = e2[d * 2 + s]; if (v > v2) { v2 = v; i2 = d; } }
    R[17 + s] = i1; R[19 + s] = i2;
    R[21 + s] = (l2[s] >= l2[2 + s]) ? 0 : 1;
  }
  int t1 = (e3[0] >= e3[1]) ? 0 : 1;
  R[23] = t1; R[24] = 1 - t1;
}

// ---------------------------------------------------------------------------
// Kernel 1b: pre-convert the selected modules' weights to bf16, transposed to
// [n][k] (the exact LDS layout) so the main kernel can stage them with pure
// async byte copies. W1 K padded 784 -> 800 with zeros.
// ---------------------------------------------------------------------------
__global__ void prep_kernel(const float* __restrict__ Wi1, const float* __restrict__ Wi2,
                            const int* __restrict__ R,
                            __bf16* __restrict__ W1t, __bf16* __restrict__ W2t) {
  const int s = blockIdx.x;
  const int m = R[4 + s];
  const float* W1 = Wi1 + (size_t)m * 784 * 256;
  const float* W2 = Wi2 + (size_t)m * 256 * 256;
  __bf16* o1 = W1t + (size_t)s * 256 * K1PAD;
  __bf16* o2 = W2t + (size_t)s * 256 * 256;
  for (int e = threadIdx.x; e < 256 * K1PAD; e += blockDim.x) {
    int n = e / K1PAD, k = e - n * K1PAD;
    o1[e] = (__bf16)((k < 784) ? W1[(size_t)k * 256 + n] : 0.f);
  }
  for (int e = threadIdx.x; e < 256 * 256; e += blockDim.x) {
    int n = e >> 8, k = e & 255;
    o2[e] = (__bf16)W2[(size_t)k * 256 + n];
  }
}

// 16-bit 16x32 operand fragment load per ISA 7.12.2:
// lanes 0-15: K = {0..7, 16..23}; lanes 16-31: K = {8..15, 24..31}; pairs contiguous.
__device__ __forceinline__ v16bf load_frag(const __bf16* base, int ld, int lane) {
  const __bf16* p = base + (lane & 15) * ld + ((lane >> 4) << 3);
  v16bf f;
#pragma unroll
  for (int j = 0; j < 16; ++j)
    f[j] = p[(j & 7) + ((j >> 3) << 4)];
  return f;
}

// Stage one 32-wide K chunk of B (256 cols) into Wt[n*LDW + k].
// ASYNC: 4 x b128 async copies per thread from pre-transposed bf16 (ldg pitch).
template <bool ASYNC>
__device__ __forceinline__ void stage_W(const __bf16* __restrict__ gbf, int ldg,
                                        const float* __restrict__ gf32, int kw,
                                        int k0, __bf16* dst, int tid) {
  if (ASYNC) {
#pragma unroll
    for (int t = 0; t < 4; ++t) {
      int e = tid + t * 256;
      int n = e >> 2, part = e & 3;
      uint32_t lds_off = (uint32_t)(uintptr_t)(dst + n * LDW + part * 8);
      async_b128(lds_off, gbf + (size_t)n * ldg + k0 + part * 8);
    }
  } else {
#pragma unroll
    for (int t = 0; t < 32; ++t) {
      int e = tid + t * 256;
      int k = e >> 8, n = e & 255;
      float v = (k < kw) ? gf32[(size_t)(k0 + k) * 256 + n] : 0.f;
      dst[n * LDW + k] = (__bf16)v;
    }
  }
}

// ---------------------------------------------------------------------------
// Kernel 2: per-slot fused 784->256->256->3 MLP via bf16 WMMA.
// grid = (B/128, 4 slots), 256 threads = 8 waves (2x4 wave grid, 64x64/wave).
// B operand double-buffered via async-to-LDS; A operand register-pipelined.
// ---------------------------------------------------------------------------
template <bool ASYNC>
__global__ __launch_bounds__(256)
void img_kernel(const float* __restrict__ x,
                const float* __restrict__ Wi1, const float* __restrict__ bi1,
                const float* __restrict__ Wi2, const float* __restrict__ bi2,
                const float* __restrict__ Wi3, const float* __restrict__ bi3,
                const int* __restrict__ R,
                const __bf16* __restrict__ W1t, const __bf16* __restrict__ W2t,
                float* __restrict__ ximg) {
  __shared__ __bf16 Y1[TB * LDY];        // 66048 B, Y2 aliases after sync
  __shared__ __bf16 Xa[TB * LDA];        // 8704 B
  __shared__ __bf16 Wt[2][256 * LDW];    // 2 x 20480 B (double buffer)
  __shared__ float  bias[256];
  __shared__ float  W3s[256 * 3];
  __shared__ float  b3s[3];

  const int tid  = threadIdx.x;
  const int lane = tid & 31;
  const int wave = tid >> 5;
  const int rowbase = (wave >> 2) * 64;
  const int colbase = (wave & 3) * 64;
  const int s  = blockIdx.y;
  const int b0 = blockIdx.x * TB;
  const int cs = R[s];
  const int ms = R[4 + s];

  const float*  Xg  = x + (size_t)cs * 784;
  const float*  W1f = Wi1 + (size_t)ms * 784 * 256;
  const float*  W2f = Wi2 + (size_t)ms * 256 * 256;
  const __bf16* W1b = W1t + (size_t)s * 256 * K1PAD;
  const __bf16* W2b = W2t + (size_t)s * 256 * 256;

  if (tid < 256) bias[tid] = bi1[ms * 256 + tid];
  for (int t = tid; t < 768; t += 256) W3s[t] = Wi3[ms * 768 + t];
  if (tid < 3) b3s[tid] = bi3[ms * 3 + tid];

  v8f acc[4][4];
#pragma unroll
  for (int i = 0; i < 4; ++i)
#pragma unroll
    for (int j = 0; j < 4; ++j) {
      v8f z = {0.f, 0.f, 0.f, 0.f, 0.f, 0.f, 0.f, 0.f};
      acc[i][j] = z;
    }

  // ---------------- GEMM1: [128 x 784(800)] x [784 x 256] ----------------
  float xr[16];
#pragma unroll
  for (int t = 0; t < 16; ++t) {               // preload X chunk 0
    int e = tid + t * 256;
    int r = e >> 5, k = e & 31;
    xr[t] = Xg[(size_t)(b0 + r) * 6272 + k];
  }
  if (ASYNC) stage_W<true>(W1b, K1PAD, nullptr, 32, 0, Wt[0], tid);  // B chunk 0

  for (int kc = 0; kc < 25; ++kc) {
    const int k0  = kc * 32;
    const int cur = ASYNC ? (kc & 1) : 0;
    __syncthreads();
#pragma unroll
    for (int t = 0; t < 16; ++t) {             // commit X chunk kc
      int e = tid + t * 256;
      int r = e >> 5, k = e & 31;
      Xa[r * LDA + k] = (__bf16)xr[t];
    }
    if (ASYNC) {
      if (kc + 1 < 25) stage_W<true>(W1b, K1PAD, nullptr, 32, k0 + 32, Wt[cur ^ 1], tid);
    } else {
      stage_W<false>(nullptr, 0, W1f, (kc == 24) ? 16 : 32, k0, Wt[0], tid);
    }
    if (kc + 1 < 25) {                          // preload X chunk kc+1 (hidden by WMMA)
      const int kn = k0 + 32;
      const int kw = (kc + 1 == 24) ? 16 : 32;
#pragma unroll
      for (int t = 0; t < 16; ++t) {
        int e = tid + t * 256;
        int r = e >> 5, k = e & 31;
        xr[t] = (k < kw) ? Xg[(size_t)(b0 + r) * 6272 + kn + k] : 0.f;
      }
      __builtin_prefetch(&Xg[(size_t)(b0 + (tid >> 5)) * 6272 + kn + 32], 0, 1);
    }
    if (ASYNC) { if (kc + 1 < 25) wait_async4(); else wait_async0(); }
    __syncthreads();
    v16bf af[4], bfr[4];
#pragma unroll
    for (int i = 0; i < 4; ++i) af[i]  = load_frag(&Xa[(rowbase + i * 16) * LDA], LDA, lane);
#pragma unroll
    for (int j = 0; j < 4; ++j) bfr[j] = load_frag(&Wt[cur][(colbase + j * 16) * LDW], LDW, lane);
#pragma unroll
    for (int i = 0; i < 4; ++i)
#pragma unroll
      for (int j = 0; j < 4; ++j)
        acc[i][j] = __builtin_amdgcn_wmma_f32_16x16x32_bf16(
            false, af[i], false, bfr[j], (short)0, acc[i][j], false, false);
  }

  // epilogue 1: relu(acc + b1) -> Y1 (bf16)
  __syncthreads();
  {
    const int hi = (lane >> 4) & 1;
    const int nl = lane & 15;
#pragma unroll
    for (int i = 0; i < 4; ++i)
#pragma unroll
      for (int j = 0; j < 4; ++j) {
        int n = colbase + j * 16 + nl;
#pragma unroll
        for (int r = 0; r < 8; ++r) {
          int m = rowbase + i * 16 + hi * 8 + r;
          float v = acc[i][j][r] + bias[n];
          Y1[m * LDY + n] = (__bf16)fmaxf(v, 0.f);
          acc[i][j][r] = 0.f;
        }
      }
  }
  __syncthreads();
  if (tid < 256) bias[tid] = bi2[ms * 256 + tid];
  if (ASYNC) stage_W<true>(W2b, 256, nullptr, 32, 0, Wt[0], tid);    // GEMM2 chunk 0

  // ---------------- GEMM2: [128 x 256] x [256 x 256] ----------------
  for (int kc = 0; kc < 8; ++kc) {
    const int k0  = kc * 32;
    const int cur = ASYNC ? (kc & 1) : 0;
    __syncthreads();
    if (ASYNC) {
      if (kc + 1 < 8) { stage_W<true>(W2b, 256, nullptr, 32, k0 + 32, Wt[cur ^ 1], tid); wait_async4(); }
      else wait_async0();
    } else {
      stage_W<false>(nullptr, 0, W2f, 32, k0, Wt[0], tid);
    }
    __syncthreads();
    v16bf af[4], bfr[4];
#pragma unroll
    for (int i = 0; i < 4; ++i) af[i]  = load_frag(&Y1[(rowbase + i * 16) * LDY + k0], LDY, lane);
#pragma unroll
    for (int j = 0; j < 4; ++j) bfr[j] = load_frag(&Wt[cur][(colbase + j * 16) * LDW], LDW, lane);
#pragma unroll
    for (int i = 0; i < 4; ++i)
#pragma unroll
      for (int j = 0; j < 4; ++j)
        acc[i][j] = __builtin_amdgcn_wmma_f32_16x16x32_bf16(
            false, af[i], false, bfr[j], (short)0, acc[i][j], false, false);
  }

  // epilogue 2: relu(acc + b2) -> Y2 (aliases Y1 storage, safe after sync)
  __syncthreads();
  {
    const int hi = (lane >> 4) & 1;
    const int nl = lane & 15;
#pragma unroll
    for (int i = 0; i < 4; ++i)
#pragma unroll
      for (int j = 0; j < 4; ++j) {
        int n = colbase + j * 16 + nl;
#pragma unroll
        for (int r = 0; r < 8; ++r) {
          int m = rowbase + i * 16 + hi * 8 + r;
          float v = acc[i][j][r] + bias[n];
          Y1[m * LDY + n] = (__bf16)fmaxf(v, 0.f);
        }
      }
  }
  __syncthreads();

  // ---------------- GEMM3: [128 x 256] x [256 x 3] (VALU dots) ----------------
  for (int e = tid; e < TB * 3; e += 256) {
    int row = e / 3, o = e - row * 3;
    float sum = b3s[o];
    const __bf16* yr = &Y1[row * LDY];
#pragma unroll 8
    for (int n = 0; n < 256; ++n) sum += (float)yr[n] * W3s[n * 3 + o];
    ximg[(size_t)(b0 + row) * 12 + s * 3 + o] = sum;
  }
}

// ---------------------------------------------------------------------------
// Kernel 3: scalar routing levels + sigmoid, one row per thread
// ---------------------------------------------------------------------------
__global__ void scalar_kernel(const float* __restrict__ ximg,
                              const float* __restrict__ Wm1, const float* __restrict__ bm1,
                              const float* __restrict__ Wm2, const float* __restrict__ bm2,
                              const int* __restrict__ R, float* __restrict__ out, int B) {
  int b = blockIdx.x * blockDim.x + threadIdx.x;
  if (b >= B) return;
  float xv[12];
#pragma unroll
  for (int i = 0; i < 12; ++i) xv[i] = ximg[(size_t)b * 12 + i];

  float x3[3];
#pragma unroll
  for (int s = 0; s < 3; ++s) {
    float v1 = xv[R[8 + s]], v2 = xv[R[11 + s]];
    int m = R[14 + s];
    float o = bm2[m];
#pragma unroll
    for (int j = 0; j < 12; ++j) {
      float h = v1 * Wm1[(m * 2 + 0) * 12 + j] + v2 * Wm1[(m * 2 + 1) * 12 + j] + bm1[m * 12 + j];
      o += fmaxf(h, 0.f) * Wm2[m * 12 + j];
    }
    x3[s] = o;
  }
  float x2[2];
#pragma unroll
  for (int s = 0; s < 2; ++s) {
    float v1 = x3[R[17 + s]], v2 = x3[R[19 + s]];
    int m = R[21 + s];
    float o = bm2[m];
#pragma unroll
    for (int j = 0; j < 12; ++j) {
      float h = v1 * Wm1[(m * 2 + 0) * 12 + j] + v2 * Wm1[(m * 2 + 1) * 12 + j] + bm1[m * 12 + j];
      o += fmaxf(h, 0.f) * Wm2[m * 12 + j];
    }
    x2[s] = o;
  }
  float a = x2[R[23]], c = x2[R[24]];
  out[(size_t)b * 2 + 0] = 1.f / (1.f + __expf(-a));
  out[(size_t)b * 2 + 1] = 1.f / (1.f + __expf(-c));
}

extern "C" void kernel_launch(void* const* d_in, const int* in_sizes, int n_in,
                              void* d_out, int out_size, void* d_ws, size_t ws_size,
                              hipStream_t stream) {
  const float* x   = (const float*)d_in[0];
  const float* e0  = (const float*)d_in[1];
  const float* l0  = (const float*)d_in[2];
  const float* e1  = (const float*)d_in[3];
  const float* l1  = (const float*)d_in[4];
  const float* e2  = (const float*)d_in[5];
  const float* l2  = (const float*)d_in[6];
  const float* e3  = (const float*)d_in[7];
  const float* Wi1 = (const float*)d_in[8];
  const float* bi1 = (const float*)d_in[9];
  const float* Wi2 = (const float*)d_in[10];
  const float* bi2 = (const float*)d_in[11];
  const float* Wi3 = (const float*)d_in[12];
  const float* bi3 = (const float*)d_in[13];
  const float* Wm1 = (const float*)d_in[14];
  const float* bm1 = (const float*)d_in[15];
  const float* Wm2 = (const float*)d_in[16];
  const float* bm2 = (const float*)d_in[17];
  float* out = (float*)d_out;

  const int B = 8192;
  const size_t off_ximg = 256;
  const size_t off_W1t  = (off_ximg + (size_t)B * 12 * 4 + 255) & ~(size_t)255;
  const size_t off_W2t  = (off_W1t + (size_t)4 * 256 * K1PAD * 2 + 255) & ~(size_t)255;
  const size_t need     = off_W2t + (size_t)4 * 256 * 256 * 2;

  int*    R    = (int*)d_ws;
  float*  ximg = (float*)((char*)d_ws + off_ximg);
  __bf16* W1t  = (__bf16*)((char*)d_ws + off_W1t);
  __bf16* W2t  = (__bf16*)((char*)d_ws + off_W2t);

  route_kernel<<<1, 32, 0, stream>>>(e0, l0, e1, l1, e2, l2, e3, R);
  dim3 g(B / TB, 4);
  if (ws_size >= need) {
    prep_kernel<<<4, 1024, 0, stream>>>(Wi1, Wi2, R, W1t, W2t);
    img_kernel<true><<<g, 256, 0, stream>>>(x, Wi1, bi1, Wi2, bi2, Wi3, bi3, R, W1t, W2t, ximg);
  } else {
    img_kernel<false><<<g, 256, 0, stream>>>(x, Wi1, bi1, Wi2, bi2, Wi3, bi3, R, W1t, W2t, ximg);
  }
  scalar_kernel<<<(B + 255) / 256, 256, 0, stream>>>(ximg, Wm1, bm1, Wm2, bm2, R, out, B);
}
